// AttentionLayer_5634997092689
// MI455X (gfx1250) — compile-verified, matrix operand order checked by reading, not run on hardware
//
#include <hip/hip_runtime.h>

typedef __bf16 bf16;
typedef __attribute__((ext_vector_type(16))) __bf16 v16bf;
typedef __attribute__((ext_vector_type(8)))  __bf16 v8bf;
typedef __attribute__((ext_vector_type(8)))  float  v8f;

#define B_  8
#define T_  256
#define M_  64
#define D_  512
#define H_  8
#define DK_ 64
#define BT_ (B_ * T_)
#define N_  (BT_ * M_)

// LDS strides (elements). All chosen so row offsets are 16B-aligned (b128
// LDS loads) and rows rotate across the 64 banks without conflicts.
#define XS_ 520   // 64 rows of x / ctx: +8 pad  -> 4-bank rotation per row
#define KS_ 72    // 64x64 K / V^T tiles: +8 pad -> 36-bank rotation per row
#define WS_ 40    // 64x32 weight k-slab: +8 pad -> 20-bank rotation per row

// ---------------------------------------------------------------------------
// gfx1250 async global->LDS copies (ASYNCcnt path). The builtin's pointer
// params are typed as 4xi32-vector pointers in addrspace(1)/(3).
// Guarded so the file compiles on toolchains without the builtins.
// ---------------------------------------------------------------------------
#if defined(__gfx1250__) &&                                              \
    __has_builtin(__builtin_amdgcn_global_load_async_to_lds_b128) &&     \
    __has_builtin(__builtin_amdgcn_s_wait_asynccnt)
#define ASYNC_LDS 1
typedef int v4i_t __attribute__((vector_size(4 * sizeof(int))));
typedef __attribute__((address_space(1))) v4i_t* gv4p;
typedef __attribute__((address_space(3))) v4i_t* lv4p;
__device__ __forceinline__ void cp16_async(const bf16* gsrc, bf16* ldst) {
  __builtin_amdgcn_global_load_async_to_lds_b128(
      (gv4p)(v4i_t*)(void*)(bf16*)gsrc, (lv4p)(v4i_t*)(void*)ldst, 0, 0);
}
__device__ __forceinline__ void async_fence() {
  __builtin_amdgcn_s_wait_asynccnt(0);
}
#else
#define ASYNC_LDS 0
__device__ __forceinline__ void cp16_async(const bf16* gsrc, bf16* ldst) {
  *(v8bf*)ldst = *(const v8bf*)gsrc;
}
__device__ __forceinline__ void async_fence() {}
#endif

// ---------------------------------------------------------------------------
// Fragment helpers (wave32):
//   A-fragment (16 rows x 32 k): lane%16 -> row,    lane/16 -> 16-elem K half
//   B-fragment (32 k x 16 cols): lane%16 -> column, lane/16 -> 16-elem K half
// Both are "lane reads 16 contiguous bf16 from its row of a row-major tile".
// ---------------------------------------------------------------------------
__device__ __forceinline__ v8f wmma_bf16(v16bf a, v16bf b, v8f c) {
  return __builtin_amdgcn_wmma_f32_16x16x32_bf16(
      false, a, false, b, (short)0, c, false, false);
}

__device__ __forceinline__ v16bf frag_bf(const bf16* p0, int stride, int lane) {
  const bf16* p = p0 + (size_t)(lane & 15) * stride + ((lane >> 4) << 4);
  return *(const v16bf*)p;
}

// Cooperative stage of one 64x32 bf16 weight k-slab into LDS (128 threads,
// 32 B per thread as two 16 B async chunks).
__device__ __forceinline__ void stage_slab(bf16* dst, const bf16* W, int kk,
                                           int tid) {
  const int row  = tid >> 1;      // 0..63 (weight row == output column)
  const int half = (tid & 1) << 4;
  const bf16* s = W + (size_t)row * D_ + kk * 32 + half;
  bf16*       d = dst + row * WS_ + half;
  cp16_async(s, d);
  cp16_async(s + 8, d + 8);
}

// ---------------------------------------------------------------------------
// Kernel 1: convert the four DxD f32 weight matrices to bf16 in workspace.
// ---------------------------------------------------------------------------
__global__ __launch_bounds__(256) void wconv_kernel(
    const float* __restrict__ Wq, const float* __restrict__ Wk,
    const float* __restrict__ Wv, const float* __restrict__ Wo,
    bf16* __restrict__ wb) {
  size_t base = ((size_t)blockIdx.x * blockDim.x + threadIdx.x) * 4;
  int mat = (int)(base >> 18);                     // D*D == 2^18
  size_t r = base & ((size_t(1) << 18) - 1);
  const float* src = (mat == 0) ? Wq : (mat == 1) ? Wk : (mat == 2) ? Wv : Wo;
  float4 f = *(const float4*)(src + r);
  wb[base + 0] = (bf16)f.x;
  wb[base + 1] = (bf16)f.y;
  wb[base + 2] = (bf16)f.z;
  wb[base + 3] = (bf16)f.w;
}

// ---------------------------------------------------------------------------
// Kernel 2: fused QKV projection + attention for one (b, t, head).
// Grid = BT*H, 128 threads = 4 waves. Wave w owns x/query/key rows
// [16w, 16w+16). Weight k-slabs are staged into double-buffered LDS with
// async global->LDS copies overlapped with the WMMA stream.
// ---------------------------------------------------------------------------
__global__ __launch_bounds__(128) void qkv_attn_kernel(
    const float* __restrict__ x, const bf16* __restrict__ wb,
    const float* __restrict__ bq, const float* __restrict__ bk,
    const float* __restrict__ bv, bf16* __restrict__ ctx) {
  extern __shared__ bf16 smem[];
  bf16* xw   = smem;                    // [4][16][XS_]  staged x (bf16)
  bf16* ks   = xw + 4 * 16 * XS_;       // [64][KS_]     K  (key-major)
  bf16* vsT  = ks + 64 * KS_;           // [64][KS_]     V^T (d-major)
  bf16* qs   = vsT + 64 * KS_;          // [4][16][KS_]  per-wave Q/P staging
  bf16* slab = qs + 4 * 16 * KS_;       // [2 buf][2 mat][64][WS_]

  const int tid  = threadIdx.x;
  const int lane = tid & 31;
  const int wave = tid >> 5;
  const int h    = blockIdx.x & (H_ - 1);
  const int bt   = blockIdx.x >> 3;

  const bf16* Wqh = wb + (size_t)0 * D_ * D_ + (size_t)(h * DK_) * D_;
  const bf16* Wkh = wb + (size_t)1 * D_ * D_ + (size_t)(h * DK_) * D_;
  const bf16* Wvh = wb + (size_t)2 * D_ * D_ + (size_t)(h * DK_) * D_;

  const int n   = lane & 15;
  const int mhi = (lane >> 4) << 3;

  bf16* xwv = xw + wave * (16 * XS_);
  bf16* qsw = qs + wave * (16 * KS_);
  auto slabp = [&](int buf, int mat) {
    return slab + ((buf * 2 + mat) * 64) * WS_;
  };

  // ---- Stage this wave's 16 rows of x as bf16 (one f32 read, ever) ----
  {
    const float* xrow = x + ((size_t)bt * M_ + wave * 16) * D_;
    for (int it = lane; it < 16 * 64; it += 32) {  // 8-float chunks
      const int row = it >> 6;
      const int c8  = (it & 63) << 3;
      const float* p = xrow + (size_t)row * D_ + c8;
      float4 f0 = *(const float4*)p;
      float4 f1 = *(const float4*)(p + 4);
      v8bf o;
      o[0] = (bf16)f0.x; o[1] = (bf16)f0.y; o[2] = (bf16)f0.z; o[3] = (bf16)f0.w;
      o[4] = (bf16)f1.x; o[5] = (bf16)f1.y; o[6] = (bf16)f1.z; o[7] = (bf16)f1.w;
      *(v8bf*)(xwv + row * XS_ + c8) = o;
    }
  }

  // ---- Phase A: K and V projections for this wave's 16 key rows ----
  v8f accK[4] = {}, accV[4] = {};
  stage_slab(slabp(0, 0), Wkh, 0, tid);
  stage_slab(slabp(0, 1), Wvh, 0, tid);
  async_fence();
  __syncthreads();
  for (int kk = 0; kk < D_ / 32; ++kk) {
    const int cur = kk & 1;
    if (kk + 1 < D_ / 32) {
      stage_slab(slabp(cur ^ 1, 0), Wkh, kk + 1, tid);
      stage_slab(slabp(cur ^ 1, 1), Wvh, kk + 1, tid);
    }
    v16bf a = frag_bf(xwv + kk * 32, XS_, lane);   // reused by 8 WMMAs
    v16bf bK[4], bV[4];
#pragma unroll
    for (int t = 0; t < 4; ++t) {
      bK[t] = frag_bf(slabp(cur, 0) + t * 16 * WS_, WS_, lane);
      bV[t] = frag_bf(slabp(cur, 1) + t * 16 * WS_, WS_, lane);
    }
#pragma unroll
    for (int t = 0; t < 4; ++t) {
      accK[t] = wmma_bf16(a, bK[t], accK[t]);
      accV[t] = wmma_bf16(a, bV[t], accV[t]);
    }
    async_fence();     // k+1 slabs resident before the tile barrier
    __syncthreads();
  }
#pragma unroll
  for (int t = 0; t < 4; ++t) {
    const float biasK = bk[h * DK_ + t * 16 + n];
    const float biasV = bv[h * DK_ + t * 16 + n];
#pragma unroll
    for (int r = 0; r < 8; ++r) {
      const int m = wave * 16 + r + mhi;   // key index
      const int c = t * 16 + n;            // dk index
      ks[m * KS_ + c]  = (bf16)(accK[t][r] + biasK);
      vsT[c * KS_ + m] = (bf16)(accV[t][r] + biasV);
    }
  }
  __syncthreads();   // ks/vsT visible to all waves; slabs free for reuse

  // ---- Phase B: Q projection (bias + 1/sqrt(DK) fused), staged to LDS ----
  v8f qacc[4] = {};
  stage_slab(slabp(0, 0), Wqh, 0, tid);
  async_fence();
  __syncthreads();
  for (int kk = 0; kk < D_ / 32; ++kk) {
    const int cur = kk & 1;
    if (kk + 1 < D_ / 32) stage_slab(slabp(cur ^ 1, 0), Wqh, kk + 1, tid);
    v16bf a = frag_bf(xwv + kk * 32, XS_, lane);
    v16bf b[4];
#pragma unroll
    for (int t = 0; t < 4; ++t)
      b[t] = frag_bf(slabp(cur, 0) + t * 16 * WS_, WS_, lane);
#pragma unroll
    for (int t = 0; t < 4; ++t) qacc[t] = wmma_bf16(a, b[t], qacc[t]);
    async_fence();
    __syncthreads();
  }
#pragma unroll
  for (int t = 0; t < 4; ++t) {
    const float bias = bq[h * DK_ + t * 16 + n];
#pragma unroll
    for (int r = 0; r < 8; ++r)
      qsw[(r + mhi) * KS_ + t * 16 + n] =
          (bf16)((qacc[t][r] + bias) * 0.125f);
  }

  // ---- Scores S = Qs @ K^T (16 x 64) ----
  v8f s[4] = {};
#pragma unroll
  for (int kk = 0; kk < 2; ++kk) {
    v16bf a = frag_bf(qsw + kk * 32, KS_, lane);
    v16bf b[4];
#pragma unroll
    for (int t = 0; t < 4; ++t)
      b[t] = frag_bf(ks + (t * 16) * KS_ + kk * 32, KS_, lane);
#pragma unroll
    for (int t = 0; t < 4; ++t) s[t] = wmma_bf16(a, b[t], s[t]);
  }

  // ---- Softmax over 64 keys (rows live in (VGPR r, lane-half); columns in
  //      lane%16 across 4 tiles -> 16-lane xor reductions per half) ----
#pragma unroll
  for (int r = 0; r < 8; ++r) {
    float mx = fmaxf(fmaxf(s[0][r], s[1][r]), fmaxf(s[2][r], s[3][r]));
#pragma unroll
    for (int msk = 8; msk >= 1; msk >>= 1)
      mx = fmaxf(mx, __shfl_xor(mx, msk, 32));
    float sum = 0.f;
#pragma unroll
    for (int t = 0; t < 4; ++t) {
      float e = __expf(s[t][r] - mx);
      s[t][r] = e;
      sum += e;
    }
#pragma unroll
    for (int msk = 8; msk >= 1; msk >>= 1)
      sum += __shfl_xor(sum, msk, 32);
    const float inv = 1.0f / sum;
#pragma unroll
    for (int t = 0; t < 4; ++t) s[t][r] *= inv;
  }

  // ---- Stage P, then context = P @ V via the transposed V tile ----
#pragma unroll
  for (int t = 0; t < 4; ++t)
#pragma unroll
    for (int r = 0; r < 8; ++r)
      qsw[(r + mhi) * KS_ + t * 16 + n] = (bf16)s[t][r];

  v8f c4[4] = {};
#pragma unroll
  for (int kk = 0; kk < 2; ++kk) {
    v16bf a = frag_bf(qsw + kk * 32, KS_, lane);
    v16bf b[4];
#pragma unroll
    for (int t = 0; t < 4; ++t)
      b[t] = frag_bf(vsT + (t * 16) * KS_ + kk * 32, KS_, lane);
#pragma unroll
    for (int t = 0; t < 4; ++t) c4[t] = wmma_bf16(a, b[t], c4[t]);
  }

  bf16* crow = ctx + ((size_t)bt * M_ + wave * 16) * D_ + h * DK_;
#pragma unroll
  for (int t = 0; t < 4; ++t)
#pragma unroll
    for (int r = 0; r < 8; ++r)
      crow[(size_t)(r + mhi) * D_ + t * 16 + n] = (bf16)c4[t][r];
}

// ---------------------------------------------------------------------------
// Kernel 3: output projection  out = ctx @ Wo^T + bo  (f32 out).
// Grid = BT blocks of 64 rows; 256 threads = 8 waves, wave w owns a
// 64-column slab, accumulating all 4x4 tiles so every fragment is reused 4x.
// The ctx tile is staged in LDS once via async copies.
// ---------------------------------------------------------------------------
__global__ __launch_bounds__(256) void oproj_kernel(
    const bf16* __restrict__ ctx, const bf16* __restrict__ wo,
    const float* __restrict__ bo, float* __restrict__ out) {
  extern __shared__ bf16 cts[];          // [64][XS_]
  const int tid  = threadIdx.x;
  const int lane = tid & 31;
  const int wave = tid >> 5;
  const int rb   = blockIdx.x * M_;
  const int colBase = wave * 64;
  const int n   = lane & 15;
  const int mhi = (lane >> 4) << 3;

  // Stage the 64x512 bf16 context tile (16 B async chunks, 256 threads).
  {
    const bf16* src = ctx + (size_t)rb * D_;
    for (int chunk = tid; chunk < (M_ * D_) / 8; chunk += 256) {
      const int row = chunk >> 6;
      const int c8  = (chunk & 63) << 3;
      cp16_async(src + (size_t)row * D_ + c8, cts + row * XS_ + c8);
    }
    async_fence();
  }
  __syncthreads();

  const bf16* wcol = wo + (size_t)colBase * D_;
  v8f acc[4][4] = {};                    // [tm][t] : 128 f32 VGPRs
  for (int kk = 0; kk < D_ / 32; ++kk) {
    if (kk + 1 < D_ / 32)
      __builtin_prefetch(wcol + (size_t)(kk + 1) * 32, 0, 1);
    v16bf a0 = frag_bf(cts + 0 * 16 * XS_ + kk * 32, XS_, lane);
    v16bf a1 = frag_bf(cts + 1 * 16 * XS_ + kk * 32, XS_, lane);
    v16bf a2 = frag_bf(cts + 2 * 16 * XS_ + kk * 32, XS_, lane);
    v16bf a3 = frag_bf(cts + 3 * 16 * XS_ + kk * 32, XS_, lane);
    v16bf b[4];
#pragma unroll
    for (int t = 0; t < 4; ++t)
      b[t] = frag_bf(wcol + (size_t)(t * 16) * D_ + kk * 32, D_, lane);
#pragma unroll
    for (int t = 0; t < 4; ++t) {
      acc[0][t] = wmma_bf16(a0, b[t], acc[0][t]);
      acc[1][t] = wmma_bf16(a1, b[t], acc[1][t]);
      acc[2][t] = wmma_bf16(a2, b[t], acc[2][t]);
      acc[3][t] = wmma_bf16(a3, b[t], acc[3][t]);
    }
  }

#pragma unroll
  for (int t = 0; t < 4; ++t) {
    const float bias = bo[colBase + t * 16 + n];
#pragma unroll
    for (int tm = 0; tm < 4; ++tm)
#pragma unroll
      for (int r = 0; r < 8; ++r)
        out[((size_t)rb + tm * 16 + r + mhi) * D_ + colBase + t * 16 + n] =
            acc[tm][t][r] + bias;
  }
}

// ---------------------------------------------------------------------------
// Host side
// ---------------------------------------------------------------------------
extern "C" void kernel_launch(void* const* d_in, const int* in_sizes, int n_in,
                              void* d_out, int out_size, void* d_ws, size_t ws_size,
                              hipStream_t stream) {
  (void)in_sizes; (void)n_in; (void)out_size; (void)ws_size;
  const float* x  = (const float*)d_in[0];
  const float* Wq = (const float*)d_in[1];
  const float* bq = (const float*)d_in[2];
  const float* Wk = (const float*)d_in[3];
  const float* bk = (const float*)d_in[4];
  const float* Wv = (const float*)d_in[5];
  const float* bv = (const float*)d_in[6];
  const float* Wo = (const float*)d_in[7];
  const float* bo = (const float*)d_in[8];

  bf16* wb  = (bf16*)d_ws;                                    // 4*D*D bf16
  bf16* ctx = (bf16*)((char*)d_ws + (size_t)4 * D_ * D_ * sizeof(bf16));

  const size_t smem2 =
      (size_t)(4 * 16 * XS_ + 64 * KS_ + 64 * KS_ + 4 * 16 * KS_ +
               2 * 2 * 64 * WS_) * sizeof(bf16);              // ~112 KB
  const size_t smem3 = (size_t)(M_ * XS_) * sizeof(bf16);     // ~65 KB

  wconv_kernel<<<(4 * D_ * D_) / (256 * 4), 256, 0, stream>>>(Wq, Wk, Wv, Wo, wb);
  qkv_attn_kernel<<<BT_ * H_, 128, smem2, stream>>>(x, wb, bq, bk, bv, ctx);
  oproj_kernel<<<BT_, 256, smem3, stream>>>(ctx, wb + (size_t)3 * D_ * D_, bo,
                                            (float*)d_out);
}
